// MeMViTAttention_79242146611851
// MI455X (gfx1250) — matrix operand order repaired and checked by reading, not compiled, not run on hardware
//
#include <hip/hip_runtime.h>
#include <hip/hip_bf16.h>

#define BATCH 2
#define HEADS 8
#define DCH 96
#define DM 768
#define TT 8
#define HSZ 28
#define WSZ 28
#define QN 6272            // 8*28*28
#define HK 14
#define WK 14
#define KNN 1568           // 8*14*14
#define ATT_SCALE 0.10206207261596575f  // 96^-0.5
#define LN_EPS 1e-5f

typedef __attribute__((ext_vector_type(16))) __bf16 bf16x16;
typedef __attribute__((ext_vector_type(8)))  float  f32x8;
typedef __attribute__((ext_vector_type(4)))  unsigned int u32x4;

union ABFrag { bf16x16 bf; u32x4 u[2]; };

__device__ __forceinline__ unsigned short f2bf(float f) {
  union { float f; unsigned u; } v; v.f = f;
  unsigned r = v.u + 0x7FFFu + ((v.u >> 16) & 1u);
  return (unsigned short)(r >> 16);
}

__device__ __forceinline__ f32x8 zero8() {
  f32x8 z = {0.f, 0.f, 0.f, 0.f, 0.f, 0.f, 0.f, 0.f};
  return z;
}

__device__ __forceinline__ f32x8 wmma_bf16(ABFrag a, ABFrag b, f32x8 c) {
  return __builtin_amdgcn_wmma_f32_16x16x32_bf16(
      /*neg_a=*/false, a.bf, /*neg_b=*/false, b.bf,
      /*c_mod=*/(short)0, c, /*reuse_a=*/false, /*reuse_b=*/false);
}

// ---------------------------------------------------------------------------
// f32 -> bf16 elementwise conversion (weights)
// ---------------------------------------------------------------------------
__global__ void cvt_bf16_kernel(const float* __restrict__ in,
                                unsigned short* __restrict__ out, int n) {
  int i = blockIdx.x * blockDim.x + threadIdx.x;
  if (i < n) out[i] = f2bf(in[i]);
}

// ---------------------------------------------------------------------------
// Depthwise 3x3x3 conv (pad 1, t-stride 1) + LayerNorm over 96 channels.
// One block = one (b, h, t_out, y_out, x_out); 96 threads = channels.
// Output bf16 in [B, Lout, 768] with channel index h*96+c.
// ---------------------------------------------------------------------------
__global__ void pool_ln_kernel(const float* __restrict__ x,
                               const float* __restrict__ w,
                               const float* __restrict__ gamma,
                               const float* __restrict__ beta,
                               int sh, int sw, int Ho, int Wo,
                               unsigned short* __restrict__ out) {
  int c = threadIdx.x;  // 0..95
  int idx = blockIdx.x;
  int xo = idx % Wo; idx /= Wo;
  int yo = idx % Ho; idx /= Ho;
  int to = idx % TT; idx /= TT;
  int h  = idx % HEADS; idx /= HEADS;
  int b  = idx;

  const float* wp = w + c * 27;
  float s = 0.f;
#pragma unroll
  for (int dt = 0; dt < 3; ++dt) {
    int ti = to + dt - 1;
    if (ti < 0 || ti >= TT) continue;
#pragma unroll
    for (int dy = 0; dy < 3; ++dy) {
      int yi = yo * sh + dy - 1;
      if (yi < 0 || yi >= HSZ) continue;
#pragma unroll
      for (int dx = 0; dx < 3; ++dx) {
        int xi = xo * sw + dx - 1;
        if (xi < 0 || xi >= WSZ) continue;
        size_t nin = (size_t)ti * (HSZ * WSZ) + (size_t)yi * WSZ + xi;
        s += wp[dt * 9 + dy * 3 + dx] *
             x[((size_t)b * QN + nin) * DM + (size_t)h * DCH + c];
      }
    }
  }
  __shared__ float sm[DCH];
  sm[c] = s;
  __syncthreads();
  float mu = 0.f;
  for (int i = 0; i < DCH; ++i) mu += sm[i];
  mu *= (1.f / DCH);
  float var = 0.f;
  for (int i = 0; i < DCH; ++i) { float d = sm[i] - mu; var += d * d; }
  var *= (1.f / DCH);
  float y = (s - mu) * rsqrtf(var + LN_EPS) * gamma[c] + beta[c];
  int Lout = TT * Ho * Wo;
  int lo = to * (Ho * Wo) + yo * Wo + xo;
  out[((size_t)b * Lout + lo) * DM + (size_t)h * DCH + c] = f2bf(y);
}

// ---------------------------------------------------------------------------
// Register-blocked bf16 WMMA GEMM: Y[M,N] = A[M,K] @ Bw[N,K]^T, fp32 acc.
// One wave computes a 16x64 strip: 4 accumulators share one A fragment per
// 32-wide K step -> 10 b128 loads per 4 v_wmma (2.5 loads/WMMA).
// mode 0: bf16 out row-major (val * scale)
// mode 1: bf16 out transposed per batch: out[b][n][t], t = m % batchL
// mode 2: f32 out row-major with bias
// ---------------------------------------------------------------------------
__global__ void gemm_wmma_bf16(const unsigned short* __restrict__ A,
                               const unsigned short* __restrict__ Bw,
                               unsigned short* __restrict__ outb,
                               float* __restrict__ outf,
                               const float* __restrict__ bias,
                               int M, int N, int K, float scale,
                               int mode, int batchL) {
  int l = threadIdx.x;
  int w = blockIdx.x * 8 + threadIdx.y;
  int nBlk = N >> 6;   // 64-wide N blocks
  int tilesM = M >> 4;
  if (w >= tilesM * nBlk) return;
  int tn4 = w % nBlk;
  int tm  = w / nBlk;
  int lane15 = l & 15;
  int hi = (l >> 4) & 1;
  int kbA = hi ? 16 : 0;   // 16-bit A frag: low lanes K 0..15, high lanes 16..31
  int kbB = hi ? 16 : 0;   // 16-bit B frag: low lanes K 0..15, high lanes 16..31
  const unsigned short* arow = A + (size_t)(tm * 16 + lane15) * K;
  const unsigned short* brow[4];
#pragma unroll
  for (int f = 0; f < 4; ++f)
    brow[f] = Bw + (size_t)(tn4 * 64 + f * 16 + lane15) * K;
  f32x8 acc[4];
#pragma unroll
  for (int f = 0; f < 4; ++f) acc[f] = zero8();
  for (int k0 = 0; k0 < K; k0 += 32) {
    ABFrag a;
    a.u[0] = *(const u32x4*)(arow + k0 + kbA);
    a.u[1] = *(const u32x4*)(arow + k0 + kbA + 8);
#pragma unroll
    for (int f = 0; f < 4; ++f) {
      ABFrag b;
      b.u[0] = *(const u32x4*)(brow[f] + k0 + kbB);
      b.u[1] = *(const u32x4*)(brow[f] + k0 + kbB + 8);
      acc[f] = wmma_bf16(a, b, acc[f]);
    }
  }
  int row0 = tm * 16 + (hi ? 8 : 0);
#pragma unroll
  for (int f = 0; f < 4; ++f) {
    int col = tn4 * 64 + f * 16 + lane15;
#pragma unroll
    for (int v = 0; v < 8; ++v) {
      int m = row0 + v;
      float val = acc[f][v] * scale;
      if (mode == 2) {
        outf[(size_t)m * N + col] = val + bias[col];
      } else if (mode == 1) {
        int bb = m / batchL;
        int t = m - bb * batchL;
        outb[((size_t)bb * N + col) * batchL + t] = f2bf(val);
      } else {
        outb[(size_t)m * N + col] = f2bf(val);
      }
    }
  }
}

// ---------------------------------------------------------------------------
// Flash attention, one wave = one 16-row q tile of one (b,h).
// Q: [B, QN, 768] bf16 (pre-scaled by 1/sqrt(96)); K: [B, KNN, 768] bf16;
// Vt: [B, 768, KNN] bf16 (transposed so PV B-fragments are contiguous).
// Online softmax in f32, P relayout (C-layout -> A-layout) through LDS.
// ---------------------------------------------------------------------------
__global__ void flash_attn_kernel(const unsigned short* __restrict__ Q,
                                  const unsigned short* __restrict__ Km,
                                  const unsigned short* __restrict__ Vt,
                                  unsigned short* __restrict__ O) {
  __shared__ __align__(16) unsigned short pbuf[8][16][32];
  int l = threadIdx.x;
  int wv = threadIdx.y;
  int w = blockIdx.x * 8 + wv;
  const int tilesQ = QN / 16;  // 392
  if (w >= BATCH * HEADS * tilesQ) return;
  int tq = w % tilesQ;
  int bh = w / tilesQ;
  int h = bh % HEADS;
  int b = bh / HEADS;

  int lane15 = l & 15;
  int hi = (l >> 4) & 1;
  int kbA = hi ? 16 : 0;
  int kbB = hi ? 16 : 0;

  // Load Q fragments once: 16 rows x 96 channels = 3 A-frags.
  const unsigned short* qrow =
      Q + ((size_t)b * QN + tq * 16 + lane15) * DM + (size_t)h * DCH;
  ABFrag qa[3];
#pragma unroll
  for (int t = 0; t < 3; ++t) {
    qa[t].u[0] = *(const u32x4*)(qrow + t * 32 + kbA);
    qa[t].u[1] = *(const u32x4*)(qrow + t * 32 + kbA + 8);
  }

  f32x8 o[6];
#pragma unroll
  for (int f = 0; f < 6; ++f) o[f] = zero8();
  float mrow[8], lrow[8];
#pragma unroll
  for (int v = 0; v < 8; ++v) { mrow[v] = -3.0e38f; lrow[v] = 0.f; }

  for (int j = 0; j < KNN / 32; ++j) {  // 49 k-blocks of 32 tokens
    const unsigned short* krow0 =
        Km + ((size_t)b * KNN + j * 32 + lane15) * DM + (size_t)h * DCH;
    const unsigned short* krow1 = krow0 + (size_t)16 * DM;
    f32x8 s0 = zero8(), s1 = zero8();
#pragma unroll
    for (int t = 0; t < 3; ++t) {
      ABFrag kf;
      kf.u[0] = *(const u32x4*)(krow0 + t * 32 + kbB);
      kf.u[1] = *(const u32x4*)(krow0 + t * 32 + kbB + 8);
      s0 = wmma_bf16(qa[t], kf, s0);
      kf.u[0] = *(const u32x4*)(krow1 + t * 32 + kbB);
      kf.u[1] = *(const u32x4*)(krow1 + t * 32 + kbB + 8);
      s1 = wmma_bf16(qa[t], kf, s1);
    }
    // Online softmax. C-layout row of reg v is (v + 8*hi); cols live across
    // the 16 lanes of each half-wave -> shfl_xor masks 1,2,4,8 reduce a row.
    float alpha[8];
#pragma unroll
    for (int v = 0; v < 8; ++v) {
      float mx = fmaxf(s0[v], s1[v]);
      mx = fmaxf(mx, __shfl_xor(mx, 1, 32));
      mx = fmaxf(mx, __shfl_xor(mx, 2, 32));
      mx = fmaxf(mx, __shfl_xor(mx, 4, 32));
      mx = fmaxf(mx, __shfl_xor(mx, 8, 32));
      float mn = fmaxf(mrow[v], mx);
      float a  = __expf(mrow[v] - mn);
      float e0 = __expf(s0[v] - mn);
      float e1 = __expf(s1[v] - mn);
      float rs = e0 + e1;
      rs += __shfl_xor(rs, 1, 32);
      rs += __shfl_xor(rs, 2, 32);
      rs += __shfl_xor(rs, 4, 32);
      rs += __shfl_xor(rs, 8, 32);
      lrow[v] = lrow[v] * a + rs;
      mrow[v] = mn;
      alpha[v] = a;
      int mr = v + (hi ? 8 : 0);
      pbuf[wv][mr][lane15]      = f2bf(e0);
      pbuf[wv][mr][lane15 + 16] = f2bf(e1);
    }
#pragma unroll
    for (int f = 0; f < 6; ++f)
#pragma unroll
      for (int v = 0; v < 8; ++v) o[f][v] *= alpha[v];

    // LDS ops are in-order per wave; the explicit wait + memory clobber also
    // stops the compiler from hoisting the A-frag loads above the stores.
    asm volatile("s_wait_dscnt 0" ::: "memory");

    ABFrag pf;
    const unsigned short* prow = &pbuf[wv][lane15][0];
    pf.u[0] = *(const u32x4*)(prow + kbA);
    pf.u[1] = *(const u32x4*)(prow + kbA + 8);

    int kv = hi ? 16 : 0;
#pragma unroll
    for (int f = 0; f < 6; ++f) {
      const unsigned short* vrow =
          Vt + ((size_t)b * DM + (size_t)h * DCH + f * 16 + lane15) * KNN +
          j * 32 + kv;
      ABFrag vf;
      vf.u[0] = *(const u32x4*)(vrow);
      vf.u[1] = *(const u32x4*)(vrow + 8);
      o[f] = wmma_bf16(pf, vf, o[f]);
    }
  }

#pragma unroll
  for (int f = 0; f < 6; ++f) {
    int c = h * DCH + f * 16 + lane15;
#pragma unroll
    for (int v = 0; v < 8; ++v) {
      int m = tq * 16 + v + (hi ? 8 : 0);
      O[((size_t)b * QN + m) * DM + c] = f2bf(o[f][v] / lrow[v]);
    }
  }
}

// ---------------------------------------------------------------------------
extern "C" void kernel_launch(void* const* d_in, const int* in_sizes, int n_in,
                              void* d_out, int out_size, void* d_ws,
                              size_t ws_size, hipStream_t stream) {
  (void)in_sizes; (void)n_in; (void)out_size; (void)ws_size;
  const float* x     = (const float*)d_in[0];
  const float* pqw   = (const float*)d_in[1];
  const float* pkw   = (const float*)d_in[2];
  const float* pvw   = (const float*)d_in[3];
  const float* nqg   = (const float*)d_in[4];
  const float* nqb   = (const float*)d_in[5];
  const float* nkg   = (const float*)d_in[6];
  const float* nkb   = (const float*)d_in[7];
  const float* nvg   = (const float*)d_in[8];
  const float* nvb   = (const float*)d_in[9];
  const float* wq    = (const float*)d_in[10];
  const float* wk    = (const float*)d_in[11];
  const float* wv    = (const float*)d_in[12];
  const float* wp    = (const float*)d_in[13];
  const float* pbias = (const float*)d_in[14];
  // d_in[15..17] = thw (8,28,28) hardcoded.

  char* ws = (char*)d_ws;
  size_t off = 0;
  auto alloc = [&](size_t bytes) -> void* {
    void* p = ws + off;
    off += (bytes + 255) & ~(size_t)255;
    return p;
  };
  unsigned short* qp   = (unsigned short*)alloc((size_t)BATCH * QN  * DM * 2);
  unsigned short* kp   = (unsigned short*)alloc((size_t)BATCH * KNN * DM * 2);
  unsigned short* vp   = (unsigned short*)alloc((size_t)BATCH * KNN * DM * 2);
  unsigned short* wqb  = (unsigned short*)alloc((size_t)DM * DM * 2);
  unsigned short* wkb  = (unsigned short*)alloc((size_t)DM * DM * 2);
  unsigned short* wvb  = (unsigned short*)alloc((size_t)DM * DM * 2);
  unsigned short* wpb  = (unsigned short*)alloc((size_t)DM * DM * 2);
  unsigned short* qlin = (unsigned short*)alloc((size_t)BATCH * QN  * DM * 2);
  unsigned short* klin = (unsigned short*)alloc((size_t)BATCH * KNN * DM * 2);
  unsigned short* vt   = (unsigned short*)alloc((size_t)BATCH * DM * KNN * 2);
  unsigned short* aout = (unsigned short*)alloc((size_t)BATCH * QN  * DM * 2);

  // 1. weights -> bf16
  const int nw = DM * DM;
  cvt_bf16_kernel<<<(nw + 255) / 256, 256, 0, stream>>>(wq, wqb, nw);
  cvt_bf16_kernel<<<(nw + 255) / 256, 256, 0, stream>>>(wk, wkb, nw);
  cvt_bf16_kernel<<<(nw + 255) / 256, 256, 0, stream>>>(wv, wvb, nw);
  cvt_bf16_kernel<<<(nw + 255) / 256, 256, 0, stream>>>(wp, wpb, nw);

  // 2. depthwise conv + LN pools
  pool_ln_kernel<<<BATCH * HEADS * TT * HSZ * WSZ, DCH, 0, stream>>>(
      x, pqw, nqg, nqb, 1, 1, HSZ, WSZ, qp);
  pool_ln_kernel<<<BATCH * HEADS * TT * HK * WK, DCH, 0, stream>>>(
      x, pkw, nkg, nkb, 2, 2, HK, WK, kp);
  pool_ln_kernel<<<BATCH * HEADS * TT * HK * WK, DCH, 0, stream>>>(
      x, pvw, nvg, nvb, 2, 2, HK, WK, vp);

  // 3. linear projections (WMMA, 16x64 strip per wave)
  auto gemm = [&](const unsigned short* A, const unsigned short* Bw,
                  unsigned short* ob, float* of, const float* bias, int M,
                  int N, int K, float scale, int mode, int batchL) {
    int waves = (M / 16) * (N / 64);
    dim3 blk(32, 8);
    gemm_wmma_bf16<<<(waves + 7) / 8, blk, 0, stream>>>(
        A, Bw, ob, of, bias, M, N, K, scale, mode, batchL);
  };
  gemm(qp, wqb, qlin, nullptr, nullptr, BATCH * QN,  DM, DM, ATT_SCALE, 0, 1);
  gemm(kp, wkb, klin, nullptr, nullptr, BATCH * KNN, DM, DM, 1.0f,      0, 1);
  gemm(vp, wvb, vt,   nullptr, nullptr, BATCH * KNN, DM, DM, 1.0f,      1, KNN);

  // 4. flash attention (WMMA QK^T + PV, online softmax)
  {
    dim3 blk(32, 8);
    int waves = BATCH * HEADS * (QN / 16);  // 6272
    flash_attn_kernel<<<(waves + 7) / 8, blk, 0, stream>>>(qlin, klin, vt, aout);
  }

  // 5. output projection, f32 + bias into d_out
  gemm(aout, wpb, nullptr, (float*)d_out, pbias, BATCH * QN, DM, DM, 1.0f, 2, 1);
}
